// Attention_19121194402320
// MI455X (gfx1250) — matrix operand order repaired and checked by reading, not compiled, not run on hardware
//
#include <hip/hip_runtime.h>
#include <hip/hip_bf16.h>

typedef _Float16 v8h  __attribute__((ext_vector_type(8)));
typedef _Float16 v16h __attribute__((ext_vector_type(16)));
typedef float    v8f  __attribute__((ext_vector_type(8)));
typedef unsigned int u32x4 __attribute__((ext_vector_type(4)));
typedef int          i32x4 __attribute__((ext_vector_type(4)));
typedef int          i32x8 __attribute__((ext_vector_type(8)));

constexpr int kB = 4, kS = 1024, kE = 768, kH = 12, kD = 64;

__device__ __forceinline__ v16h cat8(v8h lo, v8h hi) {
  return __builtin_shufflevector(lo, hi, 0,1,2,3,4,5,6,7,8,9,10,11,12,13,14,15);
}

__device__ __forceinline__ v8f wmma_f16f32(v16h a, v16h b, v8f c) {
  // D = A(16x32 f16) * B(32x16 f16) + C(16x16 f32)
  return __builtin_amdgcn_wmma_f32_16x16x32_f16(false, a, false, b, (short)0, c,
                                                false, false);
}

// LDS byte address of a __shared__ object: flat LDS-aperture addresses carry
// the LDS offset in the low 32 bits (ISA 10.2 aperture mapping).
__device__ __forceinline__ unsigned int lds_addr_of(const void* p) {
  return (unsigned int)(uintptr_t)p;
}

// ---------------------------------------------------------------------------
// TDM: async 2D f16 tile load global -> LDS (CDNA5 ISA).
// tile = tile_d1 rows x tile_d0 elems, row stride stride0 elems, 2B elements.
// ---------------------------------------------------------------------------
__device__ __forceinline__ void tdm_load_2d_f16(unsigned int lds_byte,
                                                const void* gaddr,
                                                int tile_d0, int tile_d1,
                                                int tensor_d0, int tensor_d1,
                                                long long stride0) {
  const unsigned long long ga = (unsigned long long)(uintptr_t)gaddr;
  u32x4 g0;
  g0[0] = 1u;                                    // count=1, user descriptor
  g0[1] = lds_byte;                              // lds_addr
  g0[2] = (unsigned int)ga;                      // global_addr[31:0]
  g0[3] = (unsigned int)(ga >> 32) | (2u << 30); // global_addr[56:32] | type=2
  i32x8 g1;
  g1[0] = (1 << 16);                             // data_size=1 -> 2 bytes
  g1[1] = (tensor_d0 & 0xFFFF) << 16;            // tensor_dim0[15:0]
  g1[2] = ((unsigned)tensor_d0 >> 16) | ((tensor_d1 & 0xFFFF) << 16);
  g1[3] = ((unsigned)tensor_d1 >> 16) | (tile_d0 << 16);   // tile_dim0
  g1[4] = tile_d1;                               // tile_dim1 (tile_dim2=0)
  g1[5] = (int)(stride0 & 0xFFFFFFFFll);         // tensor_dim0_stride[31:0]
  g1[6] = (int)(stride0 >> 32);                  // stride0[47:32] (dim1_stride=0)
  g1[7] = 0;
  i32x4 z4 = {};
#if __clang_major__ >= 23
  i32x8 z8 = {};
  __builtin_amdgcn_tensor_load_to_lds(g0, g1, z4, z4, z8, 0);
#else
  __builtin_amdgcn_tensor_load_to_lds(g0, g1, z4, z4, 0);
#endif
}

// ---------------------------------------------------------------------------
// Projection: dst(f16)[m][n] = sum_e X[m][e] * W[n][e]   (x @ W.T)
// WG computes 128(M) x 64(N); per K-step stage X/W tiles f32->f16 in LDS,
// each wave reuses its A fragment across 4 N-tiles -> 4 WMMAs per step.
// ---------------------------------------------------------------------------
__global__ __launch_bounds__(256) void proj_f16_kernel(
    const float* __restrict__ X, const float* __restrict__ W,
    _Float16* __restrict__ dst) {
  __shared__ _Float16 sX[128][32];
  __shared__ _Float16 sW[64][32];

  const int tid = threadIdx.x;
  const int lane = tid & 31, wave = tid >> 5;
  const int lg = lane >> 4, lr = lane & 15;
  const int n0 = blockIdx.x * 64;
  const int m0 = blockIdx.y * 128;

  v8f acc[4] = {};

  // staging assignments: 2 threads per X row (16 f32 each), 4 per W row (8 f32)
  const int xr = tid >> 1, xc = (tid & 1) * 16;
  const int wr = tid >> 2, wc = (tid & 3) * 8;
  const float* Xrow = X + (size_t)(m0 + xr) * kE + xc;
  const float* Wrow = W + (size_t)(n0 + wr) * kE + wc;

  for (int e0 = 0; e0 < kE; e0 += 32) {
    __syncthreads();
    {
      v8h h0, h1, w0;
      #pragma unroll
      for (int i = 0; i < 8; ++i) h0[i] = (_Float16)Xrow[e0 + i];
      #pragma unroll
      for (int i = 0; i < 8; ++i) h1[i] = (_Float16)Xrow[e0 + 8 + i];
      #pragma unroll
      for (int i = 0; i < 8; ++i) w0[i] = (_Float16)Wrow[e0 + i];
      *(v8h*)&sX[xr][xc]     = h0;
      *(v8h*)&sX[xr][xc + 8] = h1;
      *(v8h*)&sW[wr][wc]     = w0;
    }
    __syncthreads();

    // A fragment: row = wave*16 + lr, K chunks {lg*8.., lg*8+16..}
    const _Float16* ar = &sX[wave * 16 + lr][lg * 8];
    const v16h a = cat8(*(const v8h*)ar, *(const v8h*)(ar + 16));
    #pragma unroll
    for (int nt = 0; nt < 4; ++nt) {
      const _Float16* br = &sW[nt * 16 + lr][lg * 8];
      const v16h bm = cat8(*(const v8h*)br, *(const v8h*)(br + 16));
      acc[nt] = wmma_f16f32(a, bm, acc[nt]);
    }
  }

  // C layout: lane, VGPR r -> row lg*8+r, col lr
  _Float16* dr = dst + (size_t)(m0 + wave * 16 + lg * 8) * kE + n0 + lr;
  #pragma unroll
  for (int nt = 0; nt < 4; ++nt) {
    #pragma unroll
    for (int r = 0; r < 8; ++r)
      dr[(size_t)r * kE + nt * 16] = (_Float16)acc[nt][r];
  }
}

// ---------------------------------------------------------------------------
// Fused flash attention per head (head = contiguous 1024x64 f16 slice due to
// the reference's raw reshape). 8 waves/WG share one head; each wave owns a
// 16-query tile with online softmax. K/V k-blocks of 32 staged in LDS by the
// Tensor Data Mover (async DMA, TENSORcnt).
// ---------------------------------------------------------------------------
__global__ __launch_bounds__(256) void flash_attn_kernel(
    const _Float16* __restrict__ Qb, const _Float16* __restrict__ Kb,
    const _Float16* __restrict__ Vb, float* __restrict__ out) {
  __shared__ _Float16 sK[32][64];        // K block (k x d), TDM destination
  __shared__ _Float16 sVraw[32][64];     // V block (k x d), TDM destination
  __shared__ _Float16 sVt[64][32];       // V transposed (d x k)
  __shared__ _Float16 sP[8][16][32];     // per-wave P tile (C->A transpose)

  const int tid = threadIdx.x;
  const int lane = tid & 31, wave = tid >> 5;
  const int lg = lane >> 4, lr = lane & 15;
  const int b = blockIdx.x / kH, h = blockIdx.x % kH;
  const size_t hb = (size_t)b * kS * kE + (size_t)h * kS * kD;
  const _Float16* Qh = Qb + hb;
  const _Float16* Kh = Kb + hb;
  const _Float16* Vh = Vb + hb;
  const int q0 = blockIdx.y * 128 + wave * 16;

  // Q A-fragments for d 0..31 and 32..63 (row stride kD, contiguous 16B chunks)
  v16h aq0, aq1;
  {
    const _Float16* qr = Qh + (size_t)(q0 + lr) * kD;
    aq0 = cat8(*(const v8h*)(qr + lg * 8),      *(const v8h*)(qr + lg * 8 + 16));
    aq1 = cat8(*(const v8h*)(qr + 32 + lg * 8), *(const v8h*)(qr + 32 + lg * 8 + 16));
  }

  v8f o0 = {}, o1 = {}, o2 = {}, o3 = {};   // O accum: 16 x 64 (4 d-tiles)
  float m[8], l[8];                          // per-row running max / sum
  #pragma unroll
  for (int r = 0; r < 8; ++r) { m[r] = -1e30f; l[r] = 0.0f; }

  for (int kb = 0; kb < kS; kb += 32) {
    __syncthreads();  // previous iteration's sK/sVt readers are done
    if (wave == 0) {
      // TDM: two async 2D tile DMAs (32 rows x 64 f16, row stride 64)
      tdm_load_2d_f16(lds_addr_of(&sK[0][0]),    Kh + (size_t)kb * kD,
                      kD, 32, kD, 32, kD);
      tdm_load_2d_f16(lds_addr_of(&sVraw[0][0]), Vh + (size_t)kb * kD,
                      kD, 32, kD, 32, kD);
      __builtin_amdgcn_s_wait_tensorcnt(0);
    }
    __syncthreads();  // publish TDM results to all waves

    // transpose V in LDS so PV B-fragments are contiguous
    #pragma unroll
    for (int i = 0; i < 8; ++i) {
      const int e = tid * 8 + i;
      const int vr = e >> 6, vc = e & 63;
      sVt[vc][vr] = sVraw[vr][vc];
    }
    __syncthreads();

    // S = Q * K^T for two 16-key sub-tiles (contraction over d)
    v8f s0 = {}, s1 = {};
    {
      const _Float16* k0 = &sK[lr][lg * 8];
      const _Float16* k1 = &sK[16 + lr][lg * 8];
      v16h b00 = cat8(*(const v8h*)(k0),      *(const v8h*)(k0 + 16));
      v16h b01 = cat8(*(const v8h*)(k0 + 32), *(const v8h*)(k0 + 48));
      v16h b10 = cat8(*(const v8h*)(k1),      *(const v8h*)(k1 + 16));
      v16h b11 = cat8(*(const v8h*)(k1 + 32), *(const v8h*)(k1 + 48));
      s0 = wmma_f16f32(aq0, b00, s0);
      s0 = wmma_f16f32(aq1, b01, s0);
      s1 = wmma_f16f32(aq0, b10, s1);
      s1 = wmma_f16f32(aq1, b11, s1);
    }

    // Online softmax: rows live in VGPR index r (row lg*8+r), cols across the
    // 16-lane group -> xor-shuffle reductions with masks 1,2,4,8.
    float p0[8], p1[8];
    #pragma unroll
    for (int r = 0; r < 8; ++r) {
      float t = fmaxf(s0[r], s1[r]);
      t = fmaxf(t, __shfl_xor(t, 1));
      t = fmaxf(t, __shfl_xor(t, 2));
      t = fmaxf(t, __shfl_xor(t, 4));
      t = fmaxf(t, __shfl_xor(t, 8));
      const float mn = fmaxf(m[r], t);
      const float sc = __expf(m[r] - mn);
      m[r] = mn;
      o0[r] *= sc; o1[r] *= sc; o2[r] *= sc; o3[r] *= sc;
      p0[r] = __expf(s0[r] - mn);
      p1[r] = __expf(s1[r] - mn);
      float rs = p0[r] + p1[r];
      rs += __shfl_xor(rs, 1);
      rs += __shfl_xor(rs, 2);
      rs += __shfl_xor(rs, 4);
      rs += __shfl_xor(rs, 8);
      l[r] = l[r] * sc + rs;
    }

    // C-layout -> A-layout transpose of P through LDS (f16)
    #pragma unroll
    for (int r = 0; r < 8; ++r) {
      sP[wave][lg * 8 + r][lr]      = (_Float16)p0[r];
      sP[wave][lg * 8 + r][16 + lr] = (_Float16)p1[r];
    }
    __syncthreads();

    const _Float16* pr = &sP[wave][lr][lg * 16];
    const v16h ap = cat8(*(const v8h*)(pr), *(const v8h*)(pr + 8));

    // O += P(16x32) * V(32x16) for the four 16-wide d tiles
    const int kbase = lg * 16;
    const _Float16* v0 = &sVt[lr][kbase];
    const _Float16* v1 = &sVt[16 + lr][kbase];
    const _Float16* v2 = &sVt[32 + lr][kbase];
    const _Float16* v3 = &sVt[48 + lr][kbase];
    o0 = wmma_f16f32(ap, cat8(*(const v8h*)v0, *(const v8h*)(v0 + 8)), o0);
    o1 = wmma_f16f32(ap, cat8(*(const v8h*)v1, *(const v8h*)(v1 + 8)), o1);
    o2 = wmma_f16f32(ap, cat8(*(const v8h*)v2, *(const v8h*)(v2 + 8)), o2);
    o3 = wmma_f16f32(ap, cat8(*(const v8h*)v3, *(const v8h*)(v3 + 8)), o3);
  }

  // Epilogue: normalize and scatter with the (B,H,S,D)->(B,S,H*D) transpose.
  float* orow = out + ((size_t)b * kS + (q0 + lg * 8)) * kE + h * kD + lr;
  #pragma unroll
  for (int r = 0; r < 8; ++r) {
    const float inv = 1.0f / l[r];
    float* p = orow + (size_t)r * kE;
    p[0]  = o0[r] * inv;
    p[16] = o1[r] * inv;
    p[32] = o2[r] * inv;
    p[48] = o3[r] * inv;
  }
}

extern "C" void kernel_launch(void* const* d_in, const int* in_sizes, int n_in,
                              void* d_out, int out_size, void* d_ws, size_t ws_size,
                              hipStream_t stream) {
  const float* key   = (const float*)d_in[0];
  const float* query = (const float*)d_in[1];
  const float* value = (const float*)d_in[2];
  const float* Wk    = (const float*)d_in[3];
  const float* Wq    = (const float*)d_in[4];
  const float* Wv    = (const float*)d_in[5];

  const size_t n = (size_t)kB * kS * kE;          // 3 * 6.3 MB f16 in d_ws
  _Float16* Qh = (_Float16*)d_ws;
  _Float16* Kh = Qh + n;
  _Float16* Vh = Kh + n;

  dim3 pg(kE / 64, (kB * kS) / 128, 1);           // 12 x 32 WGs, 8 waves each
  proj_f16_kernel<<<pg, 256, 0, stream>>>(query, Wq, Qh);
  proj_f16_kernel<<<pg, 256, 0, stream>>>(key,   Wk, Kh);
  proj_f16_kernel<<<pg, 256, 0, stream>>>(value, Wv, Vh);

  flash_attn_kernel<<<dim3(kB * kH, kS / 128, 1), 256, 0, stream>>>(
      Qh, Kh, Vh, (float*)d_out);

  (void)in_sizes; (void)n_in; (void)out_size; (void)ws_size;
}
